// MyRNN_34522947125671
// MI455X (gfx1250) — matrix operand order repaired and checked by reading, not compiled, not run on hardware
//
#include <hip/hip_runtime.h>
#include <math.h>

// Problem dims (match reference)
#define SEQ   512
#define BATCH 64
#define HID   1024   // IN == H == 1024
#define BH    (BATCH * HID)

typedef __bf16  v16bf __attribute__((ext_vector_type(16)));
typedef __bf16  v8bf  __attribute__((ext_vector_type(8)));
typedef float   v8f   __attribute__((ext_vector_type(8)));

// ---------------------------------------------------------------------------
// WMMA fragment loaders, contiguous k order shared by A and B (wave32).
//   A (16x32, MxK): lane l<16:  M=l,    K = 0..15  (contiguous)
//                   lane l>=16: M=l-16, K = 16..31 (contiguous)
//   B (32x16, KxN): lanes 0-15: N=lane,    K = 0..15  (contiguous)
//                   lanes 16-31: N=lane-16, K = 16..31 (contiguous)
// ---------------------------------------------------------------------------
__device__ __forceinline__ v16bf load_a_bf16(const __bf16* __restrict__ row,
                                             int kbase, int lane) {
  return *(const v16bf*)(row + kbase + ((lane & 16) ? 16 : 0));
}

__device__ __forceinline__ v16bf load_b_bf16(const __bf16* __restrict__ wrow,
                                             int kbase, int lane) {
  // wrow = &W_bf16[n * HID]; W row-major [N][K] (torch Linear weight)
  return *(const v16bf*)(wrow + kbase + ((lane & 16) ? 16 : 0));
}

// ---------------------------------------------------------------------------
// Prep kernels
// ---------------------------------------------------------------------------
__global__ void pack_weights_bf16(const float* __restrict__ Wi0,
                                  const float* __restrict__ Wh0,
                                  const float* __restrict__ Wi1,
                                  const float* __restrict__ Wh1,
                                  __bf16* __restrict__ dst) {
  size_t i = (size_t)blockIdx.x * blockDim.x + threadIdx.x;  // 4 * 2^20 elems
  int which = (int)(i >> 20);
  size_t off = i & ((size_t)HID * HID - 1);
  const float* src = (which == 0) ? Wi0 : (which == 1) ? Wh0
                    : (which == 2) ? Wi1 : Wh1;
  dst[i] = (__bf16)src[off];
}

// One streaming pass: x f32 -> bf16 (halves GEMM A-traffic, kills per-tile cvt)
__global__ void pack_x_bf16(const float* __restrict__ X, __bf16* __restrict__ Xb) {
  size_t i = ((size_t)blockIdx.x * blockDim.x + threadIdx.x) * 4;
  float4 v = *(const float4*)(X + i);
  __bf16* d = Xb + i;
  d[0] = (__bf16)v.x; d[1] = (__bf16)v.y; d[2] = (__bf16)v.z; d[3] = (__bf16)v.w;
}

__global__ void combine_bias(const float* __restrict__ bi0, const float* __restrict__ bh0,
                             const float* __restrict__ bi1, const float* __restrict__ bh1,
                             float* __restrict__ b0, float* __restrict__ b1) {
  int i = blockIdx.x * blockDim.x + threadIdx.x;
  if (i < HID) { b0[i] = bi0[i] + bh0[i]; b1[i] = bi1[i] + bh1[i]; }
}

__global__ void zero_bf16(__bf16* __restrict__ p, int n) {
  int i = blockIdx.x * blockDim.x + threadIdx.x;
  if (i < n) p[i] = (__bf16)0.0f;
}

// ---------------------------------------------------------------------------
// Batched input GEMM: Z[t][m][n] = sum_k xb[t*B+m][k] * Wi0b[n][k] + (bi0+bh0)[n]
// One wave per 16x16 tile; all-bf16 loads; f32 accumulate; writes f32 into
// d_out[t] (consumed by layer1 there, later overwritten by h2).
// ---------------------------------------------------------------------------
__global__ void z0_gemm(const __bf16* __restrict__ Xb,   // [SEQ*BATCH][HID] bf16
                        const __bf16* __restrict__ Wb,   // Wi0 bf16 [HID][HID]
                        const float* __restrict__ b0,    // [HID]
                        float* __restrict__ Z) {         // [SEQ*BATCH][HID]
  const int lane = threadIdx.x & 31;
  const int wave = blockIdx.x * (blockDim.x >> 5) + (threadIdx.x >> 5);
  const int mt = wave >> 6;    // 0..2047  (M = SEQ*BATCH = 32768)
  const int nt = wave & 63;    // 0..63    (N = 1024)

  const int m_lane = (mt << 4) + (lane & 15);
  const int ncol   = (nt << 4) + (lane & 15);

  const __bf16* arow = Xb + (size_t)m_lane * HID;
  const __bf16* brow = Wb + (size_t)ncol   * HID;

  v8f acc = {};
#pragma unroll 4
  for (int kc = 0; kc < HID; kc += 32) {
    v16bf a = load_a_bf16(arow, kc, lane);
    v16bf b = load_b_bf16(brow, kc, lane);
    acc = __builtin_amdgcn_wmma_f32_16x16x32_bf16(false, a, false, b,
                                                  (short)0, acc, false, false);
  }

  const int mbase = (mt << 4) + ((lane & 16) ? 8 : 0);
  const float bias = b0[ncol];
#pragma unroll
  for (int r = 0; r < 8; ++r)
    Z[(size_t)(mbase + r) * HID + ncol] = acc[r] + bias;
}

// ---------------------------------------------------------------------------
// Pipelined recurrence phase p (p = 0..SEQ):
//   blocks 32..63 (if do_l1): h1(p)   = tanh(Z(p) + h1(p-1) @ Wh0^T)
//   blocks  0..31 (if do_l2): h2(p-1) = tanh(h1(p-1) @ Wi1^T + h2(p-2) @ Wh1^T + b1)
// The two halves only depend on results of the PREVIOUS phase, so they run in
// one launch -> 513 dependent launches instead of 1024.
// ---------------------------------------------------------------------------
__global__ void rnn_phase(const __bf16* __restrict__ h1prev, // h1(p-1) bf16
                          __bf16* __restrict__ h1out,        // h1(p)   bf16
                          const __bf16* __restrict__ h2prev, // h2(p-2) bf16
                          __bf16* __restrict__ h2out,        // h2(p-1) bf16
                          const __bf16* __restrict__ Wh0b,
                          const __bf16* __restrict__ Wi1b,
                          const __bf16* __restrict__ Wh1b,
                          const float* __restrict__ Zt,      // Z(p) f32 (layer1)
                          const float* __restrict__ b1v,
                          float* __restrict__ out_prev,      // out[p-1] f32 (layer2)
                          float* __restrict__ hfin0,
                          float* __restrict__ hfin1,
                          int do_l1, int do_l2) {
  const int lane = threadIdx.x & 31;
  const int half = blockIdx.x >> 5;                 // 1 = layer1, 0 = layer2
  const int wave = (blockIdx.x & 31) * (blockDim.x >> 5) + (threadIdx.x >> 5);
  const int mt = wave >> 6;   // 0..3   (M = 64)
  const int nt = wave & 63;   // 0..63  (N = 1024)

  const int m_lane = (mt << 4) + (lane & 15);
  const int ncol   = (nt << 4) + (lane & 15);
  const int mbase  = (mt << 4) + ((lane & 16) ? 8 : 0);

  if (half == 1) {            // -------- layer 1: h1(p) --------
    if (!do_l1) return;
    const __bf16* arow = h1prev + (size_t)m_lane * HID;
    const __bf16* brow = Wh0b   + (size_t)ncol   * HID;
    v8f acc = {};
#pragma unroll 4
    for (int kc = 0; kc < HID; kc += 32) {
      v16bf a = load_a_bf16(arow, kc, lane);
      v16bf b = load_b_bf16(brow, kc, lane);
      acc = __builtin_amdgcn_wmma_f32_16x16x32_bf16(false, a, false, b,
                                                    (short)0, acc, false, false);
    }
#pragma unroll
    for (int r = 0; r < 8; ++r) {
      size_t idx = (size_t)(mbase + r) * HID + ncol;
      float h = tanhf(acc[r] + Zt[idx]);   // Z already holds x@Wi0^T + b0
      h1out[idx] = (__bf16)h;
      hfin0[idx] = h;                      // final write survives -> h_final[0]
    }
  } else {                    // -------- layer 2: h2(p-1) --------
    if (!do_l2) return;
    const __bf16* a1row = h1prev + (size_t)m_lane * HID;
    const __bf16* a2row = h2prev + (size_t)m_lane * HID;
    const __bf16* b1row = Wi1b   + (size_t)ncol   * HID;
    const __bf16* b2row = Wh1b   + (size_t)ncol   * HID;
    v8f acc = {};
#pragma unroll 4
    for (int kc = 0; kc < HID; kc += 32) {
      v16bf a = load_a_bf16(a1row, kc, lane);
      v16bf b = load_b_bf16(b1row, kc, lane);
      acc = __builtin_amdgcn_wmma_f32_16x16x32_bf16(false, a, false, b,
                                                    (short)0, acc, false, false);
    }
#pragma unroll 4
    for (int kc = 0; kc < HID; kc += 32) {
      v16bf a = load_a_bf16(a2row, kc, lane);
      v16bf b = load_b_bf16(b2row, kc, lane);
      acc = __builtin_amdgcn_wmma_f32_16x16x32_bf16(false, a, false, b,
                                                    (short)0, acc, false, false);
    }
    const float bias = b1v[ncol];
#pragma unroll
    for (int r = 0; r < 8; ++r) {
      size_t idx = (size_t)(mbase + r) * HID + ncol;
      float h = tanhf(acc[r] + bias);
      h2out[idx]    = (__bf16)h;
      out_prev[idx] = h;                   // outputs[p-1] = h2
      hfin1[idx]    = h;                   // final write survives -> h_final[1]
    }
  }
}

// ---------------------------------------------------------------------------
// Host driver
// ---------------------------------------------------------------------------
extern "C" void kernel_launch(void* const* d_in, const int* in_sizes, int n_in,
                              void* d_out, int out_size, void* d_ws, size_t ws_size,
                              hipStream_t stream) {
  (void)in_sizes; (void)n_in; (void)out_size; (void)ws_size;

  const float* x   = (const float*)d_in[0];
  const float* Wi0 = (const float*)d_in[1];
  const float* bi0 = (const float*)d_in[2];
  const float* Wh0 = (const float*)d_in[3];
  const float* bh0 = (const float*)d_in[4];
  const float* Wi1 = (const float*)d_in[5];
  const float* bi1 = (const float*)d_in[6];
  const float* Wh1 = (const float*)d_in[7];
  const float* bh1 = (const float*)d_in[8];

  // Workspace carve-up (~72.5 MB): bf16 weights, biases, h double-buffers, x bf16
  __bf16* Wi0b = (__bf16*)d_ws;
  __bf16* Wh0b = Wi0b + (size_t)HID * HID;
  __bf16* Wi1b = Wh0b + (size_t)HID * HID;
  __bf16* Wh1b = Wi1b + (size_t)HID * HID;
  float*  b0   = (float*)(Wh1b + (size_t)HID * HID);
  float*  b1   = b0 + HID;
  __bf16* h1buf = (__bf16*)(b1 + HID);                 // 2 x [BATCH][HID]
  __bf16* h2buf = h1buf + 2 * (size_t)BH;              // 2 x [BATCH][HID]
  __bf16* Xb    = h2buf + 2 * (size_t)BH;              // [SEQ*BATCH][HID]

  float* outs  = (float*)d_out;
  float* hfin0 = outs + (size_t)SEQ * BH;              // h_final[0]
  float* hfin1 = hfin0 + (size_t)BH;                   // h_final[1]

  // Prep: bf16 weights + x (L2/HBM streaming), fused biases, zeroed h state
  pack_weights_bf16<<<(4 * HID * HID) / 256, 256, 0, stream>>>(Wi0, Wh0, Wi1, Wh1, Wi0b);
  pack_x_bf16<<<((size_t)SEQ * BH / 4) / 256, 256, 0, stream>>>(x, Xb);
  combine_bias<<<(HID + 255) / 256, 256, 0, stream>>>(bi0, bh0, bi1, bh1, b0, b1);
  zero_bf16<<<(4 * BH) / 256, 256, 0, stream>>>(h1buf, 4 * BH);

  // Batched non-recurrent half: Z0 = x @ Wi0^T + (bi0+bh0), written into d_out.
  // M = 32768 -> 2048 m-tiles x 64 n-tiles = 131072 waves / 8 per block.
  z0_gemm<<<16384, 256, 0, stream>>>(Xb, Wi0b, b0, outs);

  // Pipelined recurrence: phase p runs layer2(p-1) || layer1(p) in one launch.
  for (int p = 0; p <= SEQ; ++p) {
    const int par = p & 1;
    const __bf16* h1p = h1buf + (size_t)(par ^ 1) * BH;   // h1(p-1)
    __bf16*       h1o = h1buf + (size_t)par       * BH;   // h1(p)
    const __bf16* h2p = h2buf + (size_t)par       * BH;   // h2(p-2)
    __bf16*       h2o = h2buf + (size_t)(par ^ 1) * BH;   // h2(p-1)
    const float*  Zt  = (p < SEQ) ? outs + (size_t)p * BH : outs;
    float*        Op  = (p >= 1) ? outs + (size_t)(p - 1) * BH : outs;

    rnn_phase<<<64, 256, 0, stream>>>(h1p, h1o, h2p, h2o,
                                      Wh0b, Wi1b, Wh1b,
                                      Zt, b1, Op, hfin0, hfin1,
                                      (int)(p < SEQ), (int)(p >= 1));
  }
}
